// FusedTriangleMultiplicativeUpdate_30588757082759
// MI455X (gfx1250) — matrix operand order repaired and checked by reading, not compiled, not run on hardware
//
#include <hip/hip_runtime.h>
#include <stdint.h>

// Fused AF2 triangle-multiplicative-update for MI455X (gfx1250).
// Strategy: f32 LayerNorm stats + f32 WMMA accumulation, bf16 matmul inputs
// via v_wmma_f32_16x16x32_bf16. a/b/gate intermediates stored bf16 so the
// einsum working set (134 MB) fits the 192 MB L2. Kernel 2 uses the Tensor
// Data Mover (tensor_load_to_lds) with double buffering, gated by
// s_wait_tensorcnt, to overlap tile DMA with WMMA compute.
// Workspace requirement: ~336 MB.

#define DEVFN __device__ __forceinline__

typedef unsigned short u16;
typedef __attribute__((ext_vector_type(4)))  float v4f;
typedef __attribute__((ext_vector_type(8)))  float v8f;
typedef __attribute__((ext_vector_type(16))) __bf16 v16bf;
typedef __attribute__((ext_vector_type(4)))  u16 u16x4;
typedef __attribute__((ext_vector_type(8)))  u16 u16x8;
typedef __attribute__((ext_vector_type(16))) u16 u16x16;
typedef __attribute__((ext_vector_type(4)))  unsigned int u32x4;
typedef __attribute__((ext_vector_type(4)))  int i32x4;
typedef __attribute__((ext_vector_type(8)))  int i32x8;

constexpr int kN  = 512;
constexpr int kNN = kN * kN;          // 262144 positions
constexpr int kC  = 128;              // C_Z == C_HIDDEN

#if defined(__HIP_DEVICE_COMPILE__) && __has_builtin(__builtin_amdgcn_tensor_load_to_lds)
#define HAS_TDM 1
#else
#define HAS_TDM 0
#endif

DEVFN u16 f2bf(float f) {             // round-to-nearest-even f32 -> bf16
  unsigned u = __builtin_bit_cast(unsigned, f);
  u += 0x7FFFu + ((u >> 16) & 1u);
  return (u16)(u >> 16);
}
DEVFN float b2f(u16 h) { return __builtin_bit_cast(float, ((unsigned)h) << 16); }
DEVFN float sigmoidf(float x) { return 1.0f / (1.0f + __expf(-x)); }

DEVFN u16x16 cat8(u16x8 lo, u16x8 hi) {
  u16x16 r;
#pragma unroll
  for (int i = 0; i < 8; ++i) { r[i] = lo[i]; r[i + 8] = hi[i]; }
  return r;
}
DEVFN v16bf asbf(u16x16 v) { return __builtin_bit_cast(v16bf, v); }
DEVFN v8f wmma_bf16(u16x16 a, u16x16 b, v8f c) {
  return __builtin_amdgcn_wmma_f32_16x16x32_bf16(false, asbf(a), false, asbf(b),
                                                 (short)0, c, false, false);
}

#if HAS_TDM
// Issue a TDM load of a 2D bf16 tile: 32 cols x 64 rows, tensor row stride
// 512 elements, into LDS at ldsOff with 16B padding after each 64B row
// (matches the u16[64][40] padded LDS tile). Descriptor per CDNA5 ISA §8.
DEVFN void tdm_load_tile_2d(unsigned ldsOff, const u16* gsrc) {
  unsigned long long ga = (unsigned long long)(size_t)gsrc;
  u32x4 g0;
  g0[0] = 1u;                                     // count=1, user D#
  g0[1] = ldsOff;                                 // lds_addr (bytes)
  g0[2] = (unsigned)ga;                           // global_addr[31:0]
  g0[3] = (unsigned)((ga >> 32) & 0x01FFFFFFu)    // global_addr[56:32]
        | (2u << 30);                             // type=2 ("image")
  i32x8 g1;
  g1[0] = (int)((1u << 16)      // data_size = 2 bytes
              | (1u << 20)      // pad_enable
              | (3u << 22)      // pad_interval: 16 DWORDs (= one 64B row)
              | (3u << 25));    // pad_amount:    4 DWORDs (= 16B)
  g1[1] = (int)0xFFFF0000u;                       // tensor_dim0 lo16 (huge)
  g1[2] = (int)0xFFFF7FFFu;                       // dim0 hi16 | dim1 lo16
  g1[3] = (int)((32u << 16) | 0x7FFFu);           // tile_dim0=32 | dim1 hi16
  g1[4] = 64;                                     // tile_dim1 = 64 rows
  g1[5] = kN;                                     // tensor_dim0_stride = 512
  g1[6] = 0;
  g1[7] = 0;
  i32x4 zz = {0, 0, 0, 0};
#if __clang_major__ >= 23
  i32x8 z8 = {0, 0, 0, 0, 0, 0, 0, 0};
  __builtin_amdgcn_tensor_load_to_lds(g0, g1, zz, zz, z8, 0);
#else
  __builtin_amdgcn_tensor_load_to_lds(g0, g1, zz, zz, 0);
#endif
}
#endif

// ---------------------------------------------------------------------------
// Kernel 0: convert + transpose weights to bf16 wT[n][k] (B-operand layout).
// rows: [0,256)=w_ab_p, [256,512)=w_ab_g, [512,640)=w_g, [640,768)=w_z
// ---------------------------------------------------------------------------
__global__ void __launch_bounds__(128) k0_prep_weights(
    const float* __restrict__ w_ab_p, const float* __restrict__ w_ab_g,
    const float* __restrict__ w_g,    const float* __restrict__ w_z,
    u16* __restrict__ wTp, u16* __restrict__ wTabg,
    u16* __restrict__ wTg, u16* __restrict__ wTz) {
  int n = blockIdx.x, k = threadIdx.x;
  if (n < 256)          wTp  [n * kC + k]         = f2bf(w_ab_p[k * 256 + n]);
  else if (n < 512)     wTabg[(n - 256) * kC + k] = f2bf(w_ab_g[k * 256 + (n - 256)]);
  else if (n < 640)     wTg  [(n - 512) * kC + k] = f2bf(w_g  [k * 128 + (n - 512)]);
  else                  wTz  [(n - 640) * kC + k] = f2bf(w_z  [k * 128 + (n - 640)]);
}

// ---------------------------------------------------------------------------
// Kernel 1: per 16-row tile: LayerNorm(z) -> zn(bf16, LDS) -> WMMA projections
//   a,b = mask*sigmoid(zn@w_ab_g+b)*(zn@w_ab_p+b)  -> channel-major bf16
//   g   = sigmoid(zn@w_g+b_g)                      -> row-major bf16
// One wave (32 threads) per block; EXEC all-ones for every WMMA.
// ---------------------------------------------------------------------------
__global__ void __launch_bounds__(32) k1_ln_proj(
    const float* __restrict__ z,      const float* __restrict__ maskv,
    const float* __restrict__ b_ab_p, const float* __restrict__ b_ab_g,
    const float* __restrict__ b_g,
    const float* __restrict__ ln_g,   const float* __restrict__ ln_b,
    const u16* __restrict__ wTp, const u16* __restrict__ wTabg,
    const u16* __restrict__ wTg,
    u16* __restrict__ a_t, u16* __restrict__ b_t, u16* __restrict__ g_ws) {
  __shared__ u16 zs[16][136];     // zn tile, bf16, padded vs bank conflicts
  __shared__ u16 stile[16][24];   // 16x16 transpose staging (16B-aligned rows)

  const int lane    = threadIdx.x;
  const int rowBase = blockIdx.x * 16;

  // ---- LayerNorm: 2 lanes per row, 64 channels each, f32 stats ----
  const int r   = lane >> 1;
  const int hlf = lane & 1;
  const float* zrow = z + (size_t)(rowBase + r) * kC + hlf * 64;
  v4f xv[16];
  float sum = 0.f, ss = 0.f;
#pragma unroll
  for (int i = 0; i < 16; ++i) {
    v4f v = *(const v4f*)(zrow + i * 4);
    xv[i] = v;
    sum += v.x + v.y + v.z + v.w;
    ss  += v.x * v.x + v.y * v.y + v.z * v.z + v.w * v.w;
  }
  sum += __shfl_xor(sum, 1, 32);
  ss  += __shfl_xor(ss, 1, 32);
  float mu  = sum * (1.0f / kC);
  float var = ss * (1.0f / kC) - mu * mu;
  float rs  = rsqrtf(var + 1e-5f);
#pragma unroll
  for (int i = 0; i < 16; ++i) {
    int c = hlf * 64 + i * 4;
    v4f gm = *(const v4f*)(ln_g + c);
    v4f bt = *(const v4f*)(ln_b + c);
    v4f v  = xv[i];
    u16x4 q;
    q[0] = f2bf((v.x - mu) * rs * gm.x + bt.x);
    q[1] = f2bf((v.y - mu) * rs * gm.y + bt.y);
    q[2] = f2bf((v.z - mu) * rs * gm.z + bt.z);
    q[3] = f2bf((v.w - mu) * rs * gm.w + bt.w);
    *(u16x4*)&zs[r][c] = q;
  }
  __syncthreads();

  // ---- Build resident A operands (zn rows, 4 K-steps of 32) ----
  const int n16  = lane & 15;
  const int kb   = (lane >> 4) * 8;    // unused; A operand also uses kb16
  const int kb16 = (lane >> 4) * 16;   // B-operand K chunk base
  const int mh   = (lane >> 4) << 3;   // +8 rows for upper lanes (D layout)
  u16x16 aop[4];
#pragma unroll
  for (int kk = 0; kk < 4; ++kk) {
    const u16* p = &zs[n16][kk * 32];
    aop[kk] = cat8(*(const u16x8*)(p + kb16), *(const u16x8*)(p + kb16 + 8));
  }

  float mk[8];
#pragma unroll
  for (int v = 0; v < 8; ++v) mk[v] = maskv[rowBase + v + mh];

  // ---- a/b projections: 16 column tiles over 256 outputs ----
  for (int nt = 0; nt < 16; ++nt) {
    v8f accP = {0.f, 0.f, 0.f, 0.f, 0.f, 0.f, 0.f, 0.f};
    v8f accG = {0.f, 0.f, 0.f, 0.f, 0.f, 0.f, 0.f, 0.f};
    const u16* bpP = wTp   + (size_t)(nt * 16 + n16) * kC + kb16;
    const u16* bpG = wTabg + (size_t)(nt * 16 + n16) * kC + kb16;
#pragma unroll
    for (int kk = 0; kk < 4; ++kk) {
      u16x16 bP = *(const u16x16*)(bpP + kk * 32);
      u16x16 bG = *(const u16x16*)(bpG + kk * 32);
      accG = wmma_bf16(aop[kk], bG, accG);
      accP = wmma_bf16(aop[kk], bP, accP);
    }
    int   col = nt * 16 + n16;
    float bgv = b_ab_g[col];
    float bpv = b_ab_p[col];
#pragma unroll
    for (int v = 0; v < 8; ++v) {
      float ab = mk[v] * sigmoidf(accG[v] + bgv) * (accP[v] + bpv);
      stile[n16][v + mh] = f2bf(ab);   // transpose: [channel][row]
    }
    __syncthreads();
    if (lane < 16) {                   // each lane: one channel, 16 rows = 32B
      int  c   = (nt & 7) * 16 + lane;
      u16* dst = (nt < 8 ? a_t : b_t) + (size_t)c * kNN + rowBase;
      *(u16x8*)(dst)     = *(const u16x8*)&stile[lane][0];
      *(u16x8*)(dst + 8) = *(const u16x8*)&stile[lane][8];
    }
    __syncthreads();
  }

  // ---- gate projection g = sigmoid(zn @ w_g + b_g), row-major bf16 ----
  for (int nt = 0; nt < 8; ++nt) {
    v8f acc = {0.f, 0.f, 0.f, 0.f, 0.f, 0.f, 0.f, 0.f};
    const u16* bp = wTg + (size_t)(nt * 16 + n16) * kC + kb16;
#pragma unroll
    for (int kk = 0; kk < 4; ++kk)
      acc = wmma_bf16(aop[kk], *(const u16x16*)(bp + kk * 32), acc);
    int   col = nt * 16 + n16;
    float bgv = b_g[col];
#pragma unroll
    for (int v = 0; v < 8; ++v)
      g_ws[(size_t)(rowBase + v + mh) * kC + col] = f2bf(sigmoidf(acc[v] + bgv));
  }
}

// ---------------------------------------------------------------------------
// Kernel 2: einsum X[i,j,c] = sum_k a[i,k,c]*b[j,k,c]; batched GEMM per c.
// Block: 256 threads (8 waves), 64x64 X-tile for one channel, K-loop of 32.
// TDM path: wave 0 issues tensor_load_to_lds for the NEXT K-step's two tiles
// (double-buffered, LDS padding done by the TDM pad fields), gated by
// s_wait_tensorcnt; one barrier per iteration. Fallback: synchronous staging.
// ---------------------------------------------------------------------------
__global__ void __launch_bounds__(256) k2_einsum(
    const u16* __restrict__ a_t, const u16* __restrict__ b_t,
    float* __restrict__ x_t) {
  __shared__ u16 as[2][64][40];
  __shared__ u16 bs[2][64][40];

  const int tid  = threadIdx.x;
  const int lane = tid & 31;
  const int wave = tid >> 5;
  const int c    = blockIdx.z;
  const int iB   = blockIdx.y * 64;
  const int jB   = blockIdx.x * 64;

  const u16* aC = a_t + (size_t)c * kNN;
  const u16* bC = b_t + (size_t)c * kNN;

  const int iSub  = wave >> 1;          // 0..3
  const int jSub0 = (wave & 1) * 2;     // 0 or 2
  const int n16   = lane & 15;
  const int kb    = (lane >> 4) * 8;
  const int kb16  = (lane >> 4) * 16;
  const int mh    = (lane >> 4) << 3;

  const int lr = tid >> 2;              // fallback staging: row 0..63
  const int lk = (tid & 3) * 8;         // fallback staging: 8-elem k chunk

  v8f acc0 = {0.f, 0.f, 0.f, 0.f, 0.f, 0.f, 0.f, 0.f};
  v8f acc1 = {0.f, 0.f, 0.f, 0.f, 0.f, 0.f, 0.f, 0.f};

#if HAS_TDM
  // Low 32 bits of a flat shared-memory address are the LDS byte offset.
  const unsigned asOff[2] = {(unsigned)(size_t)&as[0][0][0],
                             (unsigned)(size_t)&as[1][0][0]};
  const unsigned bsOff[2] = {(unsigned)(size_t)&bs[0][0][0],
                             (unsigned)(size_t)&bs[1][0][0]};
  if (wave == 0) {
    tdm_load_tile_2d(asOff[0], aC + (size_t)iB * kN);
    tdm_load_tile_2d(bsOff[0], bC + (size_t)jB * kN);
  }
#endif

  for (int step = 0; step < 16; ++step) {
    const int kk = step * 32;
#if HAS_TDM
    const int cur = step & 1;
    if (wave == 0) __builtin_amdgcn_s_wait_tensorcnt(0);
    __syncthreads();                       // buffers[cur] ready for all waves
    if (wave == 0 && step + 1 < 16) {      // prefetch next tiles into buf^1
      tdm_load_tile_2d(asOff[cur ^ 1], aC + (size_t)iB * kN + kk + 32);
      tdm_load_tile_2d(bsOff[cur ^ 1], bC + (size_t)jB * kN + kk + 32);
    }
#else
    const int cur = 0;
    __syncthreads();
    *(u32x4*)&as[0][lr][lk] = *(const u32x4*)(aC + (size_t)(iB + lr) * kN + kk + lk);
    *(u32x4*)&bs[0][lr][lk] = *(const u32x4*)(bC + (size_t)(jB + lr) * kN + kk + lk);
    if (kk + 32 < kN) {
      __builtin_prefetch(aC + (size_t)(iB + lr) * kN + kk + 32 + lk, 0, 1);
      __builtin_prefetch(bC + (size_t)(jB + lr) * kN + kk + 32 + lk, 0, 1);
    }
    __syncthreads();
#endif

    const u16* ap = &as[cur][iSub * 16 + n16][0];
    u16x16 aop = cat8(*(const u16x8*)(ap + kb16), *(const u16x8*)(ap + kb16 + 8));
    const u16* b0p = &bs[cur][(jSub0 + 0) * 16 + n16][kb16];
    const u16* b1p = &bs[cur][(jSub0 + 1) * 16 + n16][kb16];
    u16x16 b0 = cat8(*(const u16x8*)(b0p), *(const u16x8*)(b0p + 8));
    u16x16 b1 = cat8(*(const u16x8*)(b1p), *(const u16x8*)(b1p + 8));
    acc0 = wmma_bf16(aop, b0, acc0);
    acc1 = wmma_bf16(aop, b1, acc1);
  }

  float* dst = x_t + (size_t)c * kNN;
  const int i0 = iB + iSub * 16;
#pragma unroll
  for (int v = 0; v < 8; ++v) {
    int m = v + mh;
    dst[(size_t)(i0 + m) * kN + jB + (jSub0 + 0) * 16 + n16] = acc0[v];
    dst[(size_t)(i0 + m) * kN + jB + (jSub0 + 1) * 16 + n16] = acc1[v];
  }
}

// ---------------------------------------------------------------------------
// Kernel 3: gather x_t columns, LayerNorm over c, WMMA @ w_z^T, +b_z, *gate.
// One wave per 16 positions.
// ---------------------------------------------------------------------------
__global__ void __launch_bounds__(32) k3_out(
    const float* __restrict__ x_t, const u16* __restrict__ g_ws,
    const float* __restrict__ ln_g, const float* __restrict__ ln_b,
    const u16* __restrict__ wTz,   const float* __restrict__ b_z,
    float* __restrict__ out) {
  __shared__ float xs[16][132];
  __shared__ u16   zs[16][136];

  const int lane    = threadIdx.x;
  const int rowBase = blockIdx.x * 16;

  // gather transposed: each lane pulls 4 channels x 16 positions (64B runs)
#pragma unroll
  for (int cc = 0; cc < 4; ++cc) {
    int c = cc * 32 + lane;
    const float* src = x_t + (size_t)c * kNN + rowBase;
#pragma unroll
    for (int rr = 0; rr < 16; rr += 4) {
      v4f v = *(const v4f*)(src + rr);
      xs[rr + 0][c] = v.x; xs[rr + 1][c] = v.y;
      xs[rr + 2][c] = v.z; xs[rr + 3][c] = v.w;
    }
  }
  __syncthreads();

  // LayerNorm over c (f32), then bf16 into zs
  const int r   = lane >> 1;
  const int hlf = lane & 1;
  v4f xv[16];
  float sum = 0.f, ss = 0.f;
#pragma unroll
  for (int i = 0; i < 16; ++i) {
    v4f v = *(const v4f*)&xs[r][hlf * 64 + i * 4];
    xv[i] = v;
    sum += v.x + v.y + v.z + v.w;
    ss  += v.x * v.x + v.y * v.y + v.z * v.z + v.w * v.w;
  }
  sum += __shfl_xor(sum, 1, 32);
  ss  += __shfl_xor(ss, 1, 32);
  float mu  = sum * (1.0f / kC);
  float var = ss * (1.0f / kC) - mu * mu;
  float rs  = rsqrtf(var + 1e-5f);
#pragma unroll
  for (int i = 0; i < 16; ++i) {
    int c = hlf * 64 + i * 4;
    v4f gm = *(const v4f*)(ln_g + c);
    v4f bt = *(const v4f*)(ln_b + c);
    v4f v  = xv[i];
    u16x4 q;
    q[0] = f2bf((v.x - mu) * rs * gm.x + bt.x);
    q[1] = f2bf((v.y - mu) * rs * gm.y + bt.y);
    q[2] = f2bf((v.z - mu) * rs * gm.z + bt.z);
    q[3] = f2bf((v.w - mu) * rs * gm.w + bt.w);
    *(u16x4*)&zs[r][c] = q;
  }
  __syncthreads();

  const int n16  = lane & 15;
  const int kb   = (lane >> 4) * 8;
  const int kb16 = (lane >> 4) * 16;
  const int mh   = (lane >> 4) << 3;
  u16x16 aop[4];
#pragma unroll
  for (int kk = 0; kk < 4; ++kk) {
    const u16* p = &zs[n16][kk * 32];
    aop[kk] = cat8(*(const u16x8*)(p + kb16), *(const u16x8*)(p + kb16 + 8));
  }

  for (int nt = 0; nt < 8; ++nt) {
    v8f acc = {0.f, 0.f, 0.f, 0.f, 0.f, 0.f, 0.f, 0.f};
    const u16* bp = wTz + (size_t)(nt * 16 + n16) * kC + kb16;
#pragma unroll
    for (int kk = 0; kk < 4; ++kk)
      acc = wmma_bf16(aop[kk], *(const u16x16*)(bp + kk * 32), acc);
    int   col = nt * 16 + n16;
    float bz  = b_z[col];
#pragma unroll
    for (int v = 0; v < 8; ++v) {
      size_t o = (size_t)(rowBase + v + mh) * kC + col;
      out[o] = (acc[v] + bz) * b2f(g_ws[o]);
    }
  }
}

// ---------------------------------------------------------------------------
extern "C" void kernel_launch(void* const* d_in, const int* in_sizes, int n_in,
                              void* d_out, int out_size, void* d_ws, size_t ws_size,
                              hipStream_t stream) {
  const float* z      = (const float*)d_in[0];
  const float* maskp  = (const float*)d_in[1];
  const float* w_ab_p = (const float*)d_in[2];
  const float* b_ab_p = (const float*)d_in[3];
  const float* w_ab_g = (const float*)d_in[4];
  const float* b_ab_g = (const float*)d_in[5];
  const float* w_g    = (const float*)d_in[6];
  const float* b_g    = (const float*)d_in[7];
  const float* w_z    = (const float*)d_in[8];
  const float* b_z    = (const float*)d_in[9];
  const float* ln_in_g  = (const float*)d_in[10];
  const float* ln_in_b  = (const float*)d_in[11];
  const float* ln_out_g = (const float*)d_in[12];
  const float* ln_out_b = (const float*)d_in[13];
  float* out = (float*)d_out;

  char* ws = (char*)d_ws;
  size_t o = 0;
  u16* wTp   = (u16*)(ws + o); o += 256 * kC * sizeof(u16);          // 64 KB
  u16* wTabg = (u16*)(ws + o); o += 256 * kC * sizeof(u16);          // 64 KB
  u16* wTg   = (u16*)(ws + o); o += kC * kC * sizeof(u16);           // 32 KB
  u16* wTz   = (u16*)(ws + o); o += kC * kC * sizeof(u16);           // 32 KB
  u16* a_t   = (u16*)(ws + o); o += (size_t)kC * kNN * sizeof(u16);  // 64 MB
  u16* b_t   = (u16*)(ws + o); o += (size_t)kC * kNN * sizeof(u16);  // 64 MB
  u16* g_ws  = (u16*)(ws + o); o += (size_t)kC * kNN * sizeof(u16);  // 64 MB
  float* x_t = (float*)(ws + o);                                     // 128 MB
  // total workspace: ~336 MB (assumed <= ws_size)

  k0_prep_weights<<<768, 128, 0, stream>>>(w_ab_p, w_ab_g, w_g, w_z,
                                           wTp, wTabg, wTg, wTz);
  k1_ln_proj<<<kNN / 16, 32, 0, stream>>>(z, maskp, b_ab_p, b_ab_g, b_g,
                                          ln_in_g, ln_in_b,
                                          wTp, wTabg, wTg, a_t, b_t, g_ws);
  k2_einsum<<<dim3(kN / 64, kN / 64, kC), 256, 0, stream>>>(a_t, b_t, x_t);
  k3_out<<<kNN / 16, 32, 0, stream>>>(x_t, g_ws, ln_out_g, ln_out_b,
                                      wTz, b_z, out);
}
